// GraphormerEncoder_43731357008136
// MI455X (gfx1250) — matrix operand order repaired and checked by reading, not compiled, not run on hardware
//
#include <hip/hip_runtime.h>
#include <hip/hip_bf16.h>

// ---------------- problem constants ----------------
#define B_   32
#define N_   256
#define D_   512
#define H_   8
#define L_   6
#define DH_  64
#define FF_  2048
#define M_   (B_*N_)      // 8192 tokens
#define TD_  (3*D_)       // 1536
#define NEGB (-10000.0f)

typedef __bf16 bf16;
typedef __attribute__((ext_vector_type(16))) __bf16 v16bf;
typedef __attribute__((ext_vector_type(8)))  float  v8f;

__constant__ int c_voc_off[9] = {0,119,128,139,151,160,165,173,175};

// ---------------- CDNA5 data-movement feature detection ----------------
#if __has_builtin(__builtin_amdgcn_global_load_async_to_lds_b128)
#define HAVE_ASYNC 1
#else
#define HAVE_ASYNC 0
#endif

#if __has_builtin(__builtin_amdgcn_tensor_load_to_lds) && (__clang_major__ == 22)
#define HAVE_TDM 1
#else
#define HAVE_TDM 0
#endif

typedef __attribute__((address_space(3))) void as3_void;
typedef int v4i_gnu __attribute__((vector_size(16)));   // matches builtin param type

__device__ __forceinline__ unsigned int lds_off(void* p) {
  return (unsigned int)(uintptr_t)(as3_void*)p;
}

#if HAVE_ASYNC
__device__ __forceinline__ void async_copy16(void* lds, const void* g) {
  __builtin_amdgcn_global_load_async_to_lds_b128(
      (__attribute__((address_space(1))) v4i_gnu*)g,
      (__attribute__((address_space(3))) v4i_gnu*)lds, 0, 0);
}
#endif

#if HAVE_TDM
typedef unsigned int u32x4_t __attribute__((ext_vector_type(4)));
typedef int          i32x8_t __attribute__((ext_vector_type(8)));
typedef int          i32x4_t __attribute__((ext_vector_type(4)));

// TDM 2D tile load: 64 rows x 64 bytes, row stride = stride_bytes, dest LDS.
// D# built per CDNA5 ISA (group0 128b, group1 256b).
__device__ __forceinline__ void tdm_load_tile64x64B(unsigned int ldsoff,
                                                    const void* gsrc,
                                                    unsigned int stride_bytes) {
  unsigned long long ga = (unsigned long long)(uintptr_t)gsrc;
  u32x4_t g0;
  g0.x = 1u;                                     // count=1 (valid user descriptor)
  g0.y = ldsoff;                                 // lds_addr (bytes)
  g0.z = (unsigned int)ga;                       // global_addr[31:0]
  g0.w = ((unsigned int)(ga >> 32) & 0x01ffffffu) | (2u << 30); // addr[56:32]|type=2
  const unsigned int td0  = 0x7fffffffu;         // tensor_dim0 (huge: no OOB clip)
  const unsigned int td1  = 0x7fffffffu;         // tensor_dim1
  const unsigned int tile0 = 16u;                // 16 x 4B = 64B per row
  const unsigned int tile1 = 64u;                // 64 rows
  const unsigned int st0  = stride_bytes >> 2;   // row stride in 4B units
  i32x8_t g1;
  g1[0] = (int)(2u << 16);                       // wg_mask=0, data_size=4B, no flags
  g1[1] = (int)((td0 & 0xffffu) << 16);          // abar_addr=0 | tdim0[15:0]
  g1[2] = (int)((td0 >> 16) | ((td1 & 0xffffu) << 16));
  g1[3] = (int)((td1 >> 16) | (tile0 << 16));    // tdim1[31:16] | tile_dim0
  g1[4] = (int)(tile1);                          // tile_dim1 | tile_dim2=0
  g1[5] = (int)st0;                              // tensor_dim0_stride[31:0]
  g1[6] = 0;                                     // stride0[47:32]=0 | stride1 lo=0
  g1[7] = 0;
  i32x4_t z = {0, 0, 0, 0};
  __builtin_amdgcn_tensor_load_to_lds(g0, g1, z, z, 0);
}
#endif

// wait for previously issued tile transfers; `more` => one stage still in flight
__device__ __forceinline__ void stage_wait(bool more, int tid) {
#if HAVE_ASYNC
#if HAVE_TDM
  if (more) { asm volatile("s_wait_asynccnt 0x1" ::: "memory"); }
  else      { asm volatile("s_wait_asynccnt 0x0" ::: "memory"); }
#else
  if (more) { asm volatile("s_wait_asynccnt 0x2" ::: "memory"); }
  else      { asm volatile("s_wait_asynccnt 0x0" ::: "memory"); }
#endif
#endif
#if HAVE_TDM
  if (tid < 32) {
    if (more) __builtin_amdgcn_s_wait_tensorcnt(1);
    else      __builtin_amdgcn_s_wait_tensorcnt(0);
  }
#endif
  (void)more; (void)tid;
}

// ---------------- WMMA helpers ----------------
__device__ __forceinline__ v8f wmma_bf16(v16bf a, v16bf b, v8f c) {
  // D = A(16x32 bf16) x B(32x16 bf16) + C(16x16 f32)
  return __builtin_amdgcn_wmma_f32_16x16x32_bf16(false, a, false, b, (short)0, c, false, false);
}

// Gather a 16x32 bf16 A-fragment from a row-major LDS row pointer, following the
// contiguous k order (B matches): lane m (0..15) holds row m; lanes 0-15 carry K{0..15},
// lanes 16-31 carry K{16..31}, packed 2 values per VGPR.
__device__ __forceinline__ v16bf afrag_gather(const bf16* row, int g) {
  v16bf a;
#pragma unroll
  for (int t = 0; t < 16; ++t) {
    int k = (g << 4) + t;
    a[t] = row[k];
  }
  return a;
}

__device__ __forceinline__ float gelu_exact(float x) {
  return 0.5f * x * (1.0f + erff(x * 0.7071067811865475f));
}

__device__ __forceinline__ float block_sum256(float v, float* red) {
  int tid = threadIdx.x;
  red[tid] = v;
  __syncthreads();
#pragma unroll
  for (int s = 128; s > 0; s >>= 1) {
    if (tid < s) red[tid] += red[tid + s];
    __syncthreads();
  }
  float r = red[0];
  __syncthreads();
  return r;
}

// ---------------- weight conversion f32 -> bf16 ----------------
__global__ __launch_bounds__(256) void k_w2bf(const float* __restrict__ src,
                                              bf16* __restrict__ dst, int count) {
  for (int i = blockIdx.x * blockDim.x + threadIdx.x; i < count;
       i += gridDim.x * blockDim.x)
    dst[i] = (bf16)src[i];
}

// ---------------- embedding sum + input LN ----------------
__global__ __launch_bounds__(256) void k_embed(const int* __restrict__ x_pad,
                                               const float* __restrict__ emb,
                                               const float* __restrict__ gam,
                                               const float* __restrict__ bet,
                                               float* __restrict__ hout) {
  __shared__ float red[256];
  int t   = blockIdx.x;       // token
  int tid = threadIdx.x;
  int d0  = tid, d1 = tid + 256;
  const int* xi = x_pad + (size_t)t * 9;
  float v0 = 0.f, v1 = 0.f;
#pragma unroll
  for (int f = 0; f < 9; ++f) {
    int row = xi[f] + c_voc_off[f];
    v0 += emb[(size_t)row * D_ + d0];
    v1 += emb[(size_t)row * D_ + d1];
  }
  float mean = block_sum256(v0 + v1, red) * (1.0f / D_);
  float a0 = v0 - mean, a1 = v1 - mean;
  float var  = block_sum256(a0 * a0 + a1 * a1, red) * (1.0f / D_);
  float inv  = rsqrtf(var + 1e-5f);
  hout[(size_t)t * D_ + d0] = a0 * inv * gam[d0] + bet[d0];
  hout[(size_t)t * D_ + d1] = a1 * inv * gam[d1] + bet[d1];
}

// ---------------- LayerNorm f32 -> bf16 ----------------
__global__ __launch_bounds__(256) void k_ln_bf16(const float* __restrict__ src,
                                                 const float* __restrict__ gam,
                                                 const float* __restrict__ bet,
                                                 bf16* __restrict__ dst) {
  __shared__ float red[256];
  int t = blockIdx.x, tid = threadIdx.x;
  int d0 = tid, d1 = tid + 256;
  float v0 = src[(size_t)t * D_ + d0];
  float v1 = src[(size_t)t * D_ + d1];
  float mean = block_sum256(v0 + v1, red) * (1.0f / D_);
  float a0 = v0 - mean, a1 = v1 - mean;
  float var = block_sum256(a0 * a0 + a1 * a1, red) * (1.0f / D_);
  float inv = rsqrtf(var + 1e-5f);
  dst[(size_t)t * D_ + d0] = (bf16)(a0 * inv * gam[d0] + bet[d0]);
  dst[(size_t)t * D_ + d1] = (bf16)(a1 * inv * gam[d1] + bet[d1]);
}

// ---------------- final LayerNorm f32 -> f32 (d_out) ----------------
__global__ __launch_bounds__(256) void k_ln_out(const float* __restrict__ src,
                                                const float* __restrict__ gam,
                                                const float* __restrict__ bet,
                                                float* __restrict__ dst) {
  __shared__ float red[256];
  int t = blockIdx.x, tid = threadIdx.x;
  int d0 = tid, d1 = tid + 256;
  float v0 = src[(size_t)t * D_ + d0];
  float v1 = src[(size_t)t * D_ + d1];
  float mean = block_sum256(v0 + v1, red) * (1.0f / D_);
  float a0 = v0 - mean, a1 = v1 - mean;
  float var = block_sum256(a0 * a0 + a1 * a1, red) * (1.0f / D_);
  float inv = rsqrtf(var + 1e-5f);
  dst[(size_t)t * D_ + d0] = a0 * inv * gam[d0] + bet[d0];
  dst[(size_t)t * D_ + d1] = a1 * inv * gam[d1] + bet[d1];
}

__global__ __launch_bounds__(256) void k_mask_out(const int* __restrict__ nm,
                                                  int* __restrict__ dst) {
  int i = blockIdx.x * blockDim.x + threadIdx.x;
  if (i < M_) dst[i] = (nm[i] != 0) ? 1 : 0;
}

// ---------------- WMMA GEMM: C = act(A[M,K] @ W[Nc,K]^T + bias) (+res) ----------
// Block = 256 threads = 8 waves; 64x64 output tile; double-buffered LDS, K-step 32.
// Tile staging: activations via async-to-LDS (ASYNCcnt), weights via TDM (TENSORcnt)
// issued by wave 0; next tile DMA overlaps current tile WMMA.
template <bool GELU, bool RES>
__global__ __launch_bounds__(256) void k_gemm(const bf16* __restrict__ A,
                                              const bf16* __restrict__ W,
                                              const float* __restrict__ bias,
                                              const float* __restrict__ res,
                                              float* __restrict__ outF,
                                              bf16* __restrict__ outB,
                                              int Mr, int Nc, int K) {
  __shared__ bf16 sA[2][64][32];
  __shared__ bf16 sB[2][64][32];
  int tid  = threadIdx.x;
  int n0   = blockIdx.x * 64;
  int m0   = blockIdx.y * 64;
  int lane = tid & 31, w = tid >> 5;
  int g    = lane >> 4, ml = lane & 15;
  int rg   = w & 3, cg2 = w >> 2;
  int lr   = tid >> 2;            // 0..63 tile row to load
  int lc   = (tid & 3) << 3;      // 0,8,16,24 (8 bf16 = 16B)

  const int nsteps = K >> 5;
  v8f acc0 = {}; v8f acc1 = {};

  // ---- prologue: issue stage 0 into buffer 0 ----
  {
    const bf16* gA = A + (size_t)(m0 + lr) * K + lc;
#if HAVE_ASYNC
    async_copy16(&sA[0][lr][lc], gA);
#else
    *(uint4*)(&sA[0][lr][lc]) = *(const uint4*)gA;
#endif
#if HAVE_TDM
    if (tid < 32)
      tdm_load_tile64x64B(lds_off(&sB[0][0][0]), W + (size_t)n0 * K, (unsigned)K * 2u);
#elif HAVE_ASYNC
    async_copy16(&sB[0][lr][lc], W + (size_t)(n0 + lr) * K + lc);
#else
    *(uint4*)(&sB[0][lr][lc]) = *(const uint4*)(W + (size_t)(n0 + lr) * K + lc);
#endif
  }

  for (int s = 0; s < nsteps; ++s) {
    int cur = s & 1;
    __syncthreads();                       // everyone done computing on buffer cur^1
    bool more = (s + 1) < nsteps;
    if (more) {
      int k0 = (s + 1) << 5;
      const bf16* gA = A + (size_t)(m0 + lr) * K + k0 + lc;
#if HAVE_ASYNC
      async_copy16(&sA[cur ^ 1][lr][lc], gA);
#else
      *(uint4*)(&sA[cur ^ 1][lr][lc]) = *(const uint4*)gA;
#endif
#if HAVE_TDM
      if (tid < 32)
        tdm_load_tile64x64B(lds_off(&sB[cur ^ 1][0][0]),
                            W + (size_t)n0 * K + k0, (unsigned)K * 2u);
#elif HAVE_ASYNC
      async_copy16(&sB[cur ^ 1][lr][lc], W + (size_t)(n0 + lr) * K + k0 + lc);
#else
      *(uint4*)(&sB[cur ^ 1][lr][lc]) = *(const uint4*)(W + (size_t)(n0 + lr) * K + k0 + lc);
#endif
      __builtin_prefetch(A + (size_t)(m0 + lr) * K + k0 + 32 + lc, 0, 1);
    }
    stage_wait(more, tid);                 // current tile transfers complete (this wave)
    __syncthreads();                       // ... and all waves'

    v16bf af = afrag_gather(&sA[cur][rg * 16 + ml][0], g);
    v16bf b0 = *(const v16bf*)(&sB[cur][(cg2 * 2 + 0) * 16 + ml][g * 16]);
    v16bf b1 = *(const v16bf*)(&sB[cur][(cg2 * 2 + 1) * 16 + ml][g * 16]);
    acc0 = wmma_bf16(af, b0, acc0);
    acc1 = wmma_bf16(af, b1, acc1);
  }

#pragma unroll
  for (int r = 0; r < 8; ++r) {
    int m  = m0 + rg * 16 + g * 8 + r;
    int c0 = n0 + cg2 * 32 + ml;
    int c1 = c0 + 16;
    float v0 = acc0[r] + bias[c0];
    float v1 = acc1[r] + bias[c1];
    if (GELU) { v0 = gelu_exact(v0); v1 = gelu_exact(v1); }
    if (RES) {
      outF[(size_t)m * Nc + c0] = res[(size_t)m * Nc + c0] + v0;
      outF[(size_t)m * Nc + c1] = res[(size_t)m * Nc + c1] + v1;
    } else {
      outB[(size_t)m * Nc + c0] = (bf16)v0;
      outB[(size_t)m * Nc + c1] = (bf16)v1;
    }
  }
  (void)Mr;
}

// ---------------- fused biased attention ----------------
// grid = (N/16 query tiles, H, B); block = 256 threads (8 waves).
// Scores 16x256 via WMMA, bias (dist+bond+pad) fused into softmax, A@V via WMMA.
__global__ __launch_bounds__(256) void k_attn(const bf16* __restrict__ qkv,
                                              const int* __restrict__ nm,
                                              const int* __restrict__ dist,
                                              const int* __restrict__ bond,
                                              const float* __restrict__ dtab,
                                              const float* __restrict__ btab,
                                              bf16* __restrict__ out) {
  __shared__ bf16  sQ[16][64];          // 2 KB
  __shared__ float sS[16][256];         // 16 KB (reused as bf16 probs)
  __shared__ bf16  sV[64][256];         // 32 KB  V^T : [d][key]
  __shared__ float sO[2][16][64];       // 8 KB split-K partials
  __shared__ float sRed[16][16];        // 1 KB
  __shared__ float sDist[13 * 8];
  __shared__ float sBond[22 * 8];
  __shared__ int   sPad[256];

  int tid = threadIdx.x;
  int q0  = blockIdx.x * 16;
  int hh  = blockIdx.y;
  int b   = blockIdx.z;
  int bN  = b * N_;
  int lane = tid & 31, w = tid >> 5;
  int g = lane >> 4, ml = lane & 15;
  int hBase = hh * DH_;

  if (tid < 13 * 8) sDist[tid] = dtab[tid];
  if (tid < 22 * 8) sBond[tid] = btab[tid];
  sPad[tid] = (nm[bN + tid] == 0) ? 1 : 0;

  // stage Q tile [16 q][64 d]
  {
    int qi = tid >> 4, d4 = (tid & 15) << 2;
    *(uint2*)(&sQ[qi][d4]) =
        *(const uint2*)(qkv + (size_t)(bN + q0 + qi) * TD_ + hBase + d4);
  }
  // stage V transposed: sV[d][key]
#pragma unroll
  for (int c = 0; c < 4; ++c) {
    int tt  = c * 256 + tid;
    int key = tt >> 2;
    int dc  = (tt & 3) << 4;
    v16bf vv = *(const v16bf*)(qkv + (size_t)(bN + key) * TD_ + 2 * D_ + hBase + dc);
#pragma unroll
    for (int j = 0; j < 16; ++j) sV[dc + j][key] = vv[j];
  }
  __syncthreads();

  // ---- scores: Q(16x64) @ K^T(64x256), 2 key-tiles per wave ----
#pragma unroll
  for (int kk = 0; kk < 2; ++kk) {
    int kt = w + kk * 8;                 // key tile 0..15
    v8f sc = {};
#pragma unroll
    for (int ds = 0; ds < 2; ++ds) {
      v16bf af = afrag_gather(&sQ[ml][ds * 32], g);
      const bf16* kb = qkv + (size_t)(bN + kt * 16 + ml) * TD_ + D_ + hBase +
                       ds * 32 + g * 16;
      v16bf bf = *(const v16bf*)kb;
      sc = wmma_bf16(af, bf, sc);
    }
#pragma unroll
    for (int r = 0; r < 8; ++r) sS[g * 8 + r][kt * 16 + ml] = sc[r];
  }
  __syncthreads();

  // ---- softmax with fused bias: thread = (row i, 16-col chunk) ----
  {
    int i  = tid >> 4;
    int jc = (tid & 15) << 4;
    int padi = sPad[q0 + i];
    float e[16];
    float lmax = -3.0e38f;
#pragma unroll
    for (int t = 0; t < 16; ++t) {
      int j  = jc + t;
      int pj = sPad[j];
      float bt;
      if (padi | pj) {
        bt = NEGB;
      } else {
        size_t pidx = ((size_t)bN + q0 + i) * N_ + j;
        int dv = dist[pidx]; dv = dv < 0 ? 0 : (dv > 12 ? 12 : dv);
        int bv = bond[pidx]; bv = bv < 0 ? 0 : (bv > 21 ? 21 : bv);
        bt = sDist[dv * 8 + hh] + sBond[bv * 8 + hh];
      }
      if (pj) bt += NEGB;                 // key-padding bias on top
      e[t] = sS[i][j] * 0.125f + bt;      // scale = 1/sqrt(64)
      lmax = fmaxf(lmax, e[t]);
    }
    sRed[i][tid & 15] = lmax;
    __syncthreads();
    float rmax = -3.0e38f;
#pragma unroll
    for (int u = 0; u < 16; ++u) rmax = fmaxf(rmax, sRed[i][u]);
    __syncthreads();
    float lsum = 0.f;
#pragma unroll
    for (int t = 0; t < 16; ++t) { e[t] = __expf(e[t] - rmax); lsum += e[t]; }
    sRed[i][tid & 15] = lsum;
    __syncthreads();
    float rsum = 0.f;
#pragma unroll
    for (int u = 0; u < 16; ++u) rsum += sRed[i][u];
    float inv = 1.0f / rsum;
    // all sS reads retired before the reductions above; safe to overwrite as bf16
    bf16* sPb = (bf16*)(&sS[0][0]);
#pragma unroll
    for (int t = 0; t < 16; ++t) sPb[(size_t)i * 256 + jc + t] = (bf16)(e[t] * inv);
  }
  __syncthreads();

  // ---- O = P(16x256) @ V(256x64): wave -> (d-tile = w&3, key-half = w>>2) ----
  {
    const bf16* sPb = (const bf16*)(&sS[0][0]);
    int dt = w & 3, half = w >> 2;
    v8f oc = {};
#pragma unroll
    for (int kk = 0; kk < 4; ++kk) {
      int ko = half * 128 + kk * 32;
      v16bf af = afrag_gather(sPb + (size_t)ml * 256 + ko, g);
      v16bf bf = *(const v16bf*)(&sV[dt * 16 + ml][ko + g * 16]);
      oc = wmma_bf16(af, bf, oc);
    }
#pragma unroll
    for (int r = 0; r < 8; ++r) sO[half][g * 8 + r][dt * 16 + ml] = oc[r];
  }
  __syncthreads();

  // combine key-halves, write [B,N,D] bf16
#pragma unroll
  for (int r = 0; r < 4; ++r) {
    int e2 = tid + r * 256;
    int qi = e2 >> 6, dd = e2 & 63;
    float v = sO[0][qi][dd] + sO[1][qi][dd];
    out[(size_t)(bN + q0 + qi) * D_ + hBase + dd] = (bf16)v;
  }
}

// ---------------- host orchestration ----------------
extern "C" void kernel_launch(void* const* d_in, const int* in_sizes, int n_in,
                              void* d_out, int out_size, void* d_ws, size_t ws_size,
                              hipStream_t stream) {
  (void)in_sizes; (void)n_in; (void)out_size; (void)ws_size;

  const int*   x_pad      = (const int*)d_in[0];
  const int*   node_mask  = (const int*)d_in[1];
  const int*   dist_pad   = (const int*)d_in[2];
  const int*   bond_pad   = (const int*)d_in[3];
  const float* node_emb   = (const float*)d_in[4];
  const float* dist_tab   = (const float*)d_in[5];
  const float* bond_tab   = (const float*)d_in[6];
  const float* in_ln_g    = (const float*)d_in[7];
  const float* in_ln_b    = (const float*)d_in[8];
  const float* out_ln_g   = (const float*)d_in[9];
  const float* out_ln_b   = (const float*)d_in[10];
  const float* in_proj_w  = (const float*)d_in[11];
  const float* in_proj_b  = (const float*)d_in[12];
  const float* out_proj_w = (const float*)d_in[13];
  const float* out_proj_b = (const float*)d_in[14];
  const float* ln1_g      = (const float*)d_in[15];
  const float* ln1_b      = (const float*)d_in[16];
  const float* ff_w1      = (const float*)d_in[17];
  const float* ff_b1      = (const float*)d_in[18];
  const float* ff_w2      = (const float*)d_in[19];
  const float* ff_b2      = (const float*)d_in[20];
  const float* ln2_g      = (const float*)d_in[21];
  const float* ln2_b      = (const float*)d_in[22];

  // workspace carve-up
  char* p = (char*)d_ws;
  float* hbuf = (float*)p;  p += (size_t)M_ * D_  * sizeof(float);
  bf16*  hn   = (bf16*)p;   p += (size_t)M_ * D_  * sizeof(bf16);
  bf16*  qkv  = (bf16*)p;   p += (size_t)M_ * TD_ * sizeof(bf16);
  bf16*  att  = (bf16*)p;   p += (size_t)M_ * D_  * sizeof(bf16);
  bf16*  ffb  = (bf16*)p;   p += (size_t)M_ * FF_ * sizeof(bf16);
  bf16*  wq   = (bf16*)p;   p += (size_t)L_ * TD_ * D_  * sizeof(bf16);
  bf16*  wo   = (bf16*)p;   p += (size_t)L_ * D_  * D_  * sizeof(bf16);
  bf16*  wf1  = (bf16*)p;   p += (size_t)L_ * FF_ * D_  * sizeof(bf16);
  bf16*  wf2  = (bf16*)p;   p += (size_t)L_ * D_  * FF_ * sizeof(bf16);

  // weight conversion (deterministic, every call)
  k_w2bf<<<4096, 256, 0, stream>>>(in_proj_w,  wq,  L_ * TD_ * D_);
  k_w2bf<<<4096, 256, 0, stream>>>(out_proj_w, wo,  L_ * D_  * D_);
  k_w2bf<<<4096, 256, 0, stream>>>(ff_w1,      wf1, L_ * FF_ * D_);
  k_w2bf<<<4096, 256, 0, stream>>>(ff_w2,      wf2, L_ * D_  * FF_);

  // h = LN(embed_sum)
  k_embed<<<M_, 256, 0, stream>>>(x_pad, node_emb, in_ln_g, in_ln_b, hbuf);

  dim3 blk(256);
  for (int l = 0; l < L_; ++l) {
    // hn = LN1(h)
    k_ln_bf16<<<M_, 256, 0, stream>>>(hbuf, ln1_g + l * D_, ln1_b + l * D_, hn);
    // qkv = hn @ Wi^T + bi
    k_gemm<false, false><<<dim3(TD_ / 64, M_ / 64), blk, 0, stream>>>(
        hn, wq + (size_t)l * TD_ * D_, in_proj_b + (size_t)l * TD_,
        nullptr, nullptr, qkv, M_, TD_, D_);
    // attention with fused bias
    k_attn<<<dim3(N_ / 16, H_, B_), blk, 0, stream>>>(
        qkv, node_mask, dist_pad, bond_pad, dist_tab, bond_tab, att);
    // h = h + att @ Wo^T + bo
    k_gemm<false, true><<<dim3(D_ / 64, M_ / 64), blk, 0, stream>>>(
        att, wo + (size_t)l * D_ * D_, out_proj_b + (size_t)l * D_,
        hbuf, hbuf, nullptr, M_, D_, D_);
    // hn = LN2(h)
    k_ln_bf16<<<M_, 256, 0, stream>>>(hbuf, ln2_g + l * D_, ln2_b + l * D_, hn);
    // ffb = gelu(hn @ W1^T + b1)
    k_gemm<true, false><<<dim3(FF_ / 64, M_ / 64), blk, 0, stream>>>(
        hn, wf1 + (size_t)l * FF_ * D_, ff_b1 + (size_t)l * FF_,
        nullptr, nullptr, ffb, M_, FF_, D_);
    // h = h + ffb @ W2^T + b2
    k_gemm<false, true><<<dim3(D_ / 64, M_ / 64), blk, 0, stream>>>(
        ffb, wf2 + (size_t)l * D_ * FF_, ff_b2 + (size_t)l * D_,
        hbuf, hbuf, nullptr, M_, D_, FF_);
  }

  // out[0 : M*D] = LN_out(h) f32 ; out[M*D : M*D + M] = node_mask int32
  k_ln_out<<<M_, 256, 0, stream>>>(hbuf, out_ln_g, out_ln_b, (float*)d_out);
  k_mask_out<<<M_ / 256, 256, 0, stream>>>(node_mask,
                                           (int*)((float*)d_out + (size_t)M_ * D_));
}
